// KeypointSampler_11373073400431
// MI455X (gfx1250) — compile-verified, matrix operand order checked by reading, not run on hardware
//
#include <hip/hip_runtime.h>
#include <math.h>
#include <stdint.h>

// ---- problem constants (from reference) ----
#define WIN    8
#define B_     16
#define H_     2048
#define W_     2048
#define HC     256          // H_/WIN
#define WC     256          // W_/WIN
#define NCELL  (B_*HC*WC)   // 1,048,576

#define CELLS_PER_WAVE 16
#define TASKS   (B_*HC*(WC/CELLS_PER_WAVE))  // 65536 wave-tasks
#define BLOCKS  2048
#define WPB     8                              // waves per 256-thread block
#define TOTW    (BLOCKS*WPB)                   // 16384 waves in grid
#define ITERS   (TASKS/TOTW)                   // 4

typedef __attribute__((ext_vector_type(16))) _Float16 v16h;
typedef __attribute__((ext_vector_type(8)))  float    v8f;
typedef int v4i __attribute__((vector_size(16)));   // matches builtin param type

// ---- gfx1250 async copy-to-LDS path (guarded; falls back to direct loads) ----
#if defined(__AMDGCN__) && __has_builtin(__builtin_amdgcn_global_load_async_to_lds_b128)
#define USE_ASYNC 1
#define GAS1(p) ((__attribute__((address_space(1))) v4i*)(p))
#define LAS3(p) ((__attribute__((address_space(3))) v4i*)(p))
#else
#define USE_ASYNC 0
#endif

// ---- raw transcendentals: bare v_log_f32 / v_exp_f32 / v_rcp_f32 (no denorm
//      guard code; all inputs here are provably in normal range) ----
#if defined(__AMDGCN__) && __has_builtin(__builtin_amdgcn_logf)
__device__ __forceinline__ float rlog2(float v) { return __builtin_amdgcn_logf(v); }
__device__ __forceinline__ float rexp2(float v) { return __builtin_amdgcn_exp2f(v); }
__device__ __forceinline__ float rrcp (float v) { return __builtin_amdgcn_rcpf(v); }
#else
__device__ __forceinline__ float rlog2(float v) { return log2f(v); }
__device__ __forceinline__ float rexp2(float v) { return exp2f(v); }
__device__ __forceinline__ float rrcp (float v) { return 1.0f / v; }
#endif

#define LN2     0.69314718f
#define INVLN2  1.44269504f
#define NLNLN2  0.36651292f   // -ln(ln 2)

__device__ __forceinline__ uint32_t hash32(uint32_t v, uint32_t salt) {
    v += salt;
    v ^= v >> 16; v *= 0x7FEB352Du;
    v ^= v >> 15; v *= 0x846CA68Bu;
    v ^= v >> 16;
    return v;
}
// uniform in (0,1), never 0 or 1 (logs stay finite, always normal floats)
__device__ __forceinline__ float u01(uint32_t hv) {
    return (float)(hv >> 9) * 1.1920929e-7f + 5.9604645e-8f;
}

__global__ __launch_bounds__(256)
void keypoint_sampler_kernel(const float* __restrict__ x,
                             const float* __restrict__ mpad,
                             float* __restrict__ out) {
    const int lane = threadIdx.x & 31;
    const int wave = threadIdx.x >> 5;
    const int c    = lane & 15;   // cell within the wave's 16-cell block
    const int h    = lane >> 4;   // which half of the cell this lane owns
    const int waveGlobal = blockIdx.x * WPB + wave;

#if USE_ASYNC
    // double-buffered LDS staging for x: 2 bufs x 8 waves x 32 lanes x 8 float4 = 64 KB
    __shared__ float4 stage[2][WPB][32][8];
    auto issueAsync = [&](int it2, int buf) {
        const int wt2  = waveGlobal + it2 * TOTW;
        const int b2   = wt2 >> 12;
        const int rem2 = wt2 & 4095;
        const int cy2  = rem2 >> 4;
        const int cx2  = ((rem2 & 15) << 4) + c;
        const float* xb2 = x + ((size_t)b2 * H_ + (size_t)cy2 * WIN) * W_ + (size_t)cx2 * WIN;
#pragma unroll
        for (int g = 0; g < 4; ++g) {
            const int r = h + ((g & 1) << 1) + ((g >> 1) << 2);
            const float* gsrc = xb2 + (size_t)r * W_;
#pragma unroll
            for (int q = 0; q < 2; ++q) {
                __builtin_amdgcn_global_load_async_to_lds_b128(
                    GAS1(gsrc + q * 4),
                    LAS3(&stage[buf][wave][lane][g * 2 + q]),
                    0, 0);
            }
        }
    };
    issueAsync(0, 0);   // prologue: prime buffer 0
#endif

    // all-ones B fragment: D[m][n] = sum_k A[m][k]
    v16h onesB;
#pragma unroll
    for (int i = 0; i < 16; ++i) onesB[i] = (_Float16)1.0f;

    for (int it = 0; it < ITERS; ++it) {
        const int wt  = waveGlobal + it * TOTW;
        const int b   = wt >> 12;        // 4096 tasks per batch image
        const int rem = wt & 4095;
        const int cy  = rem >> 4;
        const int cx  = ((rem & 15) << 4) + c;

        const size_t rowBase = ((size_t)b * H_ + (size_t)cy * WIN) * W_ + (size_t)cx * WIN;
        const float* xb = x    + rowBase;
        const float* mb = mpad + rowBase;

#if USE_ASYNC
        // overlap: kick off next tile's async copies, then wait for current tile
        if (it + 1 < ITERS) {
            issueAsync(it + 1, (it + 1) & 1);
            asm volatile("s_wait_asynccnt 8" ::: "memory");  // 8 newest may remain in flight
        } else {
            asm volatile("s_wait_asynccnt 0" ::: "memory");
        }
#endif

        // prefetch next task's mask rows (global_prefetch_b8)
        if (it + 1 < ITERS) {
            const int wt2  = waveGlobal + (it + 1) * TOTW;
            const int b2   = wt2 >> 12;
            const int rem2 = wt2 & 4095;
            const int cy2  = rem2 >> 4;
            const int cx2  = ((rem2 & 15) << 4) + c;
            const float* pf = mpad + ((size_t)b2 * H_ + (size_t)cy2 * WIN) * W_ + (size_t)cx2 * WIN;
#pragma unroll
            for (int g = 0; g < 4; ++g) {
                const int r = h + ((g & 1) << 1) + ((g >> 1) << 2);
                __builtin_prefetch(pf + (size_t)r * W_, 0, 1);
            }
#if !USE_ASYNC
            const float* pfx = x + ((size_t)b2 * H_ + (size_t)cy2 * WIN) * W_ + (size_t)cx2 * WIN;
#pragma unroll
            for (int g = 0; g < 4; ++g) {
                const int r = h + ((g & 1) << 1) + ((g >> 1) << 2);
                __builtin_prefetch(pfx + (size_t)r * W_, 0, 1);
            }
#endif
        }

        // ---- gather this lane's 32 x-elements (rows h, h+2, 4+h, 6+h) ----
        float xv[32];
#if USE_ASYNC
#pragma unroll
        for (int g = 0; g < 4; ++g) {
#pragma unroll
            for (int q = 0; q < 2; ++q) {
                const float4 v = stage[it & 1][wave][lane][g * 2 + q];
                xv[g*8 + q*4 + 0] = v.x; xv[g*8 + q*4 + 1] = v.y;
                xv[g*8 + q*4 + 2] = v.z; xv[g*8 + q*4 + 3] = v.w;
            }
        }
#else
#pragma unroll
        for (int g = 0; g < 4; ++g) {
            const int r = h + ((g & 1) << 1) + ((g >> 1) << 2);
            const float4* px = (const float4*)(xb + (size_t)r * W_);
            const float4  a0 = px[0];
            const float4  a1 = px[1];
            xv[g*8+0] = a0.x; xv[g*8+1] = a0.y; xv[g*8+2] = a0.z; xv[g*8+3] = a0.w;
            xv[g*8+4] = a1.x; xv[g*8+5] = a1.y; xv[g*8+6] = a1.z; xv[g*8+7] = a1.w;
        }
#endif
        (void)xb;

        // ---- mask min, folded on the fly from direct b128 loads ----
        float mmin = 3.0e38f;
#pragma unroll
        for (int g = 0; g < 4; ++g) {
            const int r = h + ((g & 1) << 1) + ((g >> 1) << 2);
            const float4* pm = (const float4*)(mb + (size_t)r * W_);
            const float4  m0 = pm[0];
            const float4  m1 = pm[1];
            mmin = fminf(mmin, fminf(fminf(fminf(m0.x, m0.y), fminf(m0.z, m0.w)),
                                     fminf(fminf(m1.x, m1.y), fminf(m1.z, m1.w))));
        }

        // ---- Gumbel-max over this lane's half, plus running max of x ----
        // g = -ln(-ln u) = -ln2*log2(-log2(u)) - ln(ln2): two bare v_log_f32 + fma
        const uint32_t cellId = ((uint32_t)b * HC + (uint32_t)cy) * WC + (uint32_t)cx;
        float best = -3.0e38f; int bj = 0; float xAt = 0.0f; float xmax = -3.0e38f;
#pragma unroll
        for (int i = 0; i < 32; ++i) {
            const int g   = i >> 3;
            const int col = i & 7;
            const int r   = h + ((g & 1) << 1) + ((g >> 1) << 2);
            const int j   = r * WIN + col;                       // index 0..63 within cell
            const float u   = u01(hash32(cellId * 64u + (uint32_t)j, 0x243F6A88u));
            const float t   = -rlog2(u);                         // -log2(u) > 0
            const float gum = fmaf(-LN2, rlog2(t), NLNLN2);
            const float v   = xv[i] + gum;
            if (v > best) { best = v; bj = j; xAt = xv[i]; }
            xmax = fmaxf(xmax, xv[i]);
        }

        // ---- combine the two half-cell lanes (L and L^16) ----
        const float bestO = __shfl_xor(best, 16, 32);
        const int   bjO   = __shfl_xor(bj,   16, 32);
        const float xAtO  = __shfl_xor(xAt,  16, 32);
        const float xmaxO = __shfl_xor(xmax, 16, 32);
        const float mminO = __shfl_xor(mmin, 16, 32);
        if (bestO > best || (bestO == best && bjO < bj)) { best = bestO; bj = bjO; xAt = xAtO; }
        xmax = fmaxf(xmax, xmaxO);
        mmin = fminf(mmin, mminO);

        // ---- softmax denominator: 16 cells x 64-wide row-sum via two WMMAs ----
        // A-fragment layout (ISA 7.12.2, 16-bit A 16x32): the row assignment above
        // makes each lane's xv[] already K-ordered for its half. exp(x - max) <= 1.
        v16h aFrag;
        v8f  acc = {};
#pragma unroll
        for (int i = 0; i < 16; ++i)
            aFrag[i] = (_Float16)rexp2((xv[i] - xmax) * INVLN2);
        acc = __builtin_amdgcn_wmma_f32_16x16x32_f16(false, aFrag, false, onesB,
                                                     (short)0, acc, false, false);
#pragma unroll
        for (int i = 0; i < 16; ++i)
            aFrag[i] = (_Float16)rexp2((xv[16 + i] - xmax) * INVLN2);
        acc = __builtin_amdgcn_wmma_f32_16x16x32_f16(false, aFrag, false, onesB,
                                                     (short)0, acc, false, false);

        // D columns are identical; lane L<16 holds rows 0..7, lane L>=16 rows 8..15.
        const int ci = c & 7;
        float t = acc[0];
#pragma unroll
        for (int i = 1; i < 8; ++i) t = (ci == i) ? (float)acc[i] : t;
        const float tO = __shfl_xor(t, 16, 32);
        const float S  = (((c >> 3) ^ h) != 0) ? tO : t;

        const float cat_lp = (xAt - xmax) - LN2 * rlog2(S);

        // ---- Bernoulli accept + log-prob ----
        // log_sigmoid(-z) = log_sigmoid(z) - z : one exp + one log total
        const float u2  = u01(hash32(cellId, 0x9E3779B9u));
        const float en  = rexp2(-xAt * INVLN2);                 // e^-z
        const float sig = rrcp(1.0f + en);
        const float accepted = (u2 < sig) ? 1.0f : 0.0f;
        const float ls_pos = -LN2 * rlog2(1.0f + en);           // log_sigmoid(z)
        const float lp = cat_lp + ls_pos - ((accepted > 0.0f) ? 0.0f : xAt);

        // ---- outputs (one writer lane per cell) ----
        if (h == 0) {
            const size_t N   = (size_t)NCELL;
            const size_t idx = (size_t)cellId;
            // keypoints [B,Hc,Wc,2], flipped to (x,y)
            out[idx * 2 + 0] = (float)(cx * WIN + (bj & 7));
            out[idx * 2 + 1] = (float)(cy * WIN + (bj >> 3));
            out[2 * N + idx] = lp;        // log_probs
            out[3 * N + idx] = accepted;  // mask
            out[4 * N + idx] = mmin;      // mp_grid
            out[5 * N + idx] = xAt;       // logits_selected
        }
    }
}

extern "C" void kernel_launch(void* const* d_in, const int* in_sizes, int n_in,
                              void* d_out, int out_size, void* d_ws, size_t ws_size,
                              hipStream_t stream) {
    (void)in_sizes; (void)n_in; (void)d_ws; (void)ws_size; (void)out_size;
    const float* x    = (const float*)d_in[0];
    const float* mpad = (const float*)d_in[1];
    float* out = (float*)d_out;
    keypoint_sampler_kernel<<<BLOCKS, 256, 0, stream>>>(x, mpad, out);
}
